// RenderingNetwork_31318901523206
// MI455X (gfx1250) — matrix operand order repaired and checked by reading, not compiled, not run on hardware
//
#include <hip/hip_runtime.h>
#include <hip/hip_bf16.h>
#include <math.h>

typedef __attribute__((ext_vector_type(16))) _Float16 v16h;
typedef __attribute__((ext_vector_type(8)))  float    v8f;

union HFrag { v16h h; uint4 u[2]; };

#define P_TOTAL 32768
#define NPHYS   4096
#define K_NB    20
#define D0PAD   544
#define D0      517
#define DH      512

// ---------------------------------------------------------------------------
// Weight-norm prep: w[o,k] = g[o] * v[o,k] / ||v[o,:]||  (f16 output, zero pad)
// ---------------------------------------------------------------------------
__global__ void prep_w_f16(const float* __restrict__ v, const float* __restrict__ g,
                           _Float16* __restrict__ w, int K, int Kpad) {
    __shared__ float red[256];
    const int o = blockIdx.x, tid = threadIdx.x;
    float s = 0.f;
    for (int k = tid; k < K; k += 256) { float x = v[(size_t)o * K + k]; s += x * x; }
    red[tid] = s; __syncthreads();
    for (int st = 128; st > 0; st >>= 1) {
        if (tid < st) red[tid] += red[tid + st];
        __syncthreads();
    }
    const float scale = g[o] / sqrtf(red[0]);
    for (int k = tid; k < Kpad; k += 256)
        w[(size_t)o * Kpad + k] = (_Float16)(k < K ? v[(size_t)o * K + k] * scale : 0.f);
}

__global__ void prep_w_f32(const float* __restrict__ v, const float* __restrict__ g,
                           float* __restrict__ w, int K) {
    __shared__ float red[256];
    const int o = blockIdx.x, tid = threadIdx.x;
    float s = 0.f;
    for (int k = tid; k < K; k += 256) { float x = v[(size_t)o * K + k]; s += x * x; }
    red[tid] = s; __syncthreads();
    for (int st = 128; st > 0; st >>= 1) {
        if (tid < st) red[tid] += red[tid + st];
        __syncthreads();
    }
    const float scale = g[o] / sqrtf(red[0]);
    for (int k = tid; k < K; k += 256)
        w[(size_t)o * K + k] = v[(size_t)o * K + k] * scale;
}

// ---------------------------------------------------------------------------
// Ball-query + smoothing + embeddings -> X row [P, 544] (f16, padded)
// ---------------------------------------------------------------------------
__global__ void features_kernel(const float* __restrict__ points,
                                const float* __restrict__ normals,
                                const float* __restrict__ viewd,
                                const float* __restrict__ fv,
                                const float* __restrict__ phys,
                                const float* __restrict__ rays,
                                const float* __restrict__ cam,
                                _Float16* __restrict__ X) {
    __shared__ float sph[NPHYS * 3];
    for (int i = threadIdx.x; i < NPHYS * 3; i += 256) sph[i] = phys[i];
    __syncthreads();

    const int p = blockIdx.x * 256 + threadIdx.x;
    const float px = points[p * 3 + 0], py = points[p * 3 + 1], pz = points[p * 3 + 2];

    // Streaming first-K ball query. Aggregate sums only (no neighbor list).
    int   cnt = 0, num = 0;
    float wsum = 0.f, swx = 0.f, swy = 0.f, swz = 0.f;
    float s1x = 0.f, s1y = 0.f, s1z = 0.f, s2x = 0.f, s2y = 0.f, s2z = 0.f;
    for (int j = 0; j < NPHYS; ++j) {
        if (cnt >= K_NB) break;
        const float qx = sph[j * 3 + 0], qy = sph[j * 3 + 1], qz = sph[j * 3 + 2];
        const float dx = px - qx, dy = py - qy, dz = pz - qz;
        const float d2 = dx * dx + dy * dy + dz * dz;
        if (d2 < 81.f) {
            ++cnt;
            const float t = sqrtf(d2) * (1.f / 9.f);
            const float w = fmaxf(1.f - t * t * t, 0.f);
            wsum += w; swx += w * qx; swy += w * qy; swz += w * qz;
            if (d2 != 0.f) {          // nn_mask = (dists != 0)
                ++num;
                const float vx = -dx, vy = -dy, vz = -dz;   // vec = nn - p
                s1x += vx; s1y += vy; s1z += vz;
                s2x += vx * vx; s2y += vy * vy; s2z += vz * vz;
            }
        }
    }
    // Zero-padded slots contribute w(||p - 0||) with neighbor = 0.
    {
        const float tp = sqrtf(px * px + py * py + pz * pz) * (1.f / 9.f);
        const float wpad = fmaxf(1.f - tp * tp * tp, 0.f);
        wsum += (float)(K_NB - cnt) * wpad;
    }
    const float density = wsum;
    const float invw = 1.f / (wsum + 1e-12f);
    const float spx = swx * invw, spy = swy * invw, spz = swz * invw;

    const float dcx = spx - cam[0], dcy = spy - cam[1], dcz = spz - cam[2];
    const float invd = 1.f / sqrtf(dcx * dcx + dcy * dcy + dcz * dcz);
    const float sdx = dcx * invd, sdy = dcy * invd, sdz = dcz * invd;

    const float fn = (float)num, invn = 1.f / (fn + 1e-12f);
    const float mx = s1x * invn, my = s1y * invn, mz = s1z * invn;
    const float vax = (s2x - 2.f * mx * s1x + fn * mx * mx) * invn;
    const float vay = (s2y - 2.f * my * s1y + fn * my * my) * invn;
    const float vaz = (s2z - 2.f * mz * s1z + fn * mz * mz) * invn;

    const int r = p >> 7;   // reps = 128 samples per ray
    const float rx = rays[r * 3 + 0], ry = rays[r * 3 + 1], rz = rays[r * 3 + 2];

    _Float16* row = X + (size_t)p * D0PAD;
    int off = 0;
    auto put = [&](float v) { row[off++] = (_Float16)v; };
    auto embed3 = [&](float x, float y, float z, int F) {
        put(x); put(y); put(z);
        float fr = 1.f;
        for (int f = 0; f < F; ++f) {
            put(sinf(fr * x)); put(sinf(fr * y)); put(sinf(fr * z));
            put(cosf(fr * x)); put(cosf(fr * y)); put(cosf(fr * z));
            fr *= 2.f;
        }
    };
    auto embed1 = [&](float x, int F) {
        put(x);
        float fr = 1.f;
        for (int f = 0; f < F; ++f) { put(sinf(fr * x)); put(cosf(fr * x)); fr *= 2.f; }
    };

    // rendering_input = [points, view_dirs, normals, feature_vectors]
    put(px); put(py); put(pz);
    put(viewd[p * 3 + 0]); put(viewd[p * 3 + 1]); put(viewd[p * 3 + 2]);
    put(normals[p * 3 + 0]); put(normals[p * 3 + 1]); put(normals[p * 3 + 2]);
    for (int k = 0; k < 256; ++k) put(fv[(size_t)p * 256 + k]);
    embed3(px, py, pz, 10);            // 63
    embed1(density, 4);                // 9
    embed3(spx, spy, spz, 10);         // 63
    embed3(vax, vay, vaz, 10);         // 63
    embed3(rx, ry, rz, 4);             // 27
    embed3(sdx, sdy, sdz, 4);          // 27  -> off == 517
    for (; off < D0PAD; ++off) row[off] = (_Float16)0.f;
}

// ---------------------------------------------------------------------------
// WMMA GEMM:  Y[P,512] = relu(X[P,K] * W[512,K]^T + b)    (f16 in, f32 acc)
// Block = 8 waves; wave tile 32(M)x64(N); block tile 256x64.
// W tile (64 x 32 halves) double-buffered in LDS, rows padded to 40 halves.
// One barrier per k-step; all B frags batched before the 8 back-to-back wmmas.
// ---------------------------------------------------------------------------
__global__ void gemm_wmma_f16(const _Float16* __restrict__ X, int ldx,
                              const _Float16* __restrict__ W, int ldw,
                              const float* __restrict__ bias,
                              _Float16* __restrict__ Y, int ldy, int K) {
    __shared__ _Float16 ldsW[2 * 64 * 40];

    const int lane   = threadIdx.x & 31;
    const int wave   = threadIdx.x >> 5;      // 0..7 -> M sub-tile
    const int laneLo = lane & 15;
    const int laneHi = lane >> 4;
    const int mBase  = blockIdx.x * 256 + wave * 32;
    const int nBase  = blockIdx.y * 64;

    const _Float16* Arow0 = X + (size_t)(mBase + laneLo) * ldx;       // M rows 0..15
    const _Float16* Arow1 = X + (size_t)(mBase + 16 + laneLo) * ldx;  // M rows 16..31

    v8f acc[2][4] = {};

    const int wr = threadIdx.x >> 2;          // 0..63 : W row within tile
    const int wc = threadIdx.x & 3;           // 0..3  : 8-half chunk

    auto stageW = [&](int buf, int k0) {
        *(uint4*)(&ldsW[buf * 2560 + wr * 40 + wc * 8]) =
            *(const uint4*)(W + (size_t)(nBase + wr) * ldw + k0 + wc * 8);
    };

    stageW(0, 0);
    int buf = 0;
    for (int k0 = 0; k0 < K; k0 += 32) {
        __syncthreads();                      // publishes tile `buf`; guards reuse
        if (k0 + 32 < K) stageW(buf ^ 1, k0 + 32);

        // A fragments: 16x32 f16 layout (k = (e&7) + (e>>3)*16 + laneHi*8)
        HFrag a0, a1;
        {
            const _Float16* ap0 = Arow0 + k0 + laneHi * 8;
            const _Float16* ap1 = Arow1 + k0 + laneHi * 8;
            a0.u[0] = *(const uint4*)(ap0);
            a0.u[1] = *(const uint4*)(ap0 + 16);
            a1.u[0] = *(const uint4*)(ap1);
            a1.u[1] = *(const uint4*)(ap1 + 16);
        }

        // B fragments (batched): 32x16, element e -> K = k0 + (e&7) + (e>>3)*16 + laneHi*8, N = laneLo
        HFrag b[4];
#pragma unroll
        for (int nt = 0; nt < 4; ++nt) {
            const _Float16* bp = &ldsW[buf * 2560 + (nt * 16 + laneLo) * 40 + laneHi * 8];
            b[nt].u[0] = *(const uint4*)(bp);
            b[nt].u[1] = *(const uint4*)(bp + 16);
        }

#pragma unroll
        for (int nt = 0; nt < 4; ++nt)
            acc[0][nt] = __builtin_amdgcn_wmma_f32_16x16x32_f16(
                false, a0.h, false, b[nt].h, (short)0, acc[0][nt], false, false);
#pragma unroll
        for (int nt = 0; nt < 4; ++nt)
            acc[1][nt] = __builtin_amdgcn_wmma_f32_16x16x32_f16(
                false, a1.h, false, b[nt].h, (short)0, acc[1][nt], false, false);

        buf ^= 1;
    }

    // Epilogue: bias + ReLU, store f16 (lanes 0..15 contiguous in N)
#pragma unroll
    for (int mt = 0; mt < 2; ++mt) {
#pragma unroll
        for (int nt = 0; nt < 4; ++nt) {
            const int n = nBase + nt * 16 + laneLo;
            const float bv = bias[n];
#pragma unroll
            for (int r = 0; r < 8; ++r) {
                const int m = mBase + mt * 16 + laneHi * 8 + r;
                const float v = fmaxf(acc[mt][nt][r] + bv, 0.f);
                Y[(size_t)m * ldy + n] = (_Float16)v;
            }
        }
    }
}

// ---------------------------------------------------------------------------
// Final layer: out[P,3] = sigmoid(Y[P,512] * W4[3,512]^T + b4)
// ---------------------------------------------------------------------------
__global__ void final_layer(const _Float16* __restrict__ Y,
                            const float* __restrict__ W4,
                            const float* __restrict__ b4,
                            float* __restrict__ out) {
    __shared__ float w[3 * DH];
    for (int i = threadIdx.x; i < 3 * DH; i += 256) w[i] = W4[i];
    __syncthreads();
    const int p = blockIdx.x * 256 + threadIdx.x;
    const _Float16* yp = Y + (size_t)p * DH;
    float a0 = b4[0], a1 = b4[1], a2 = b4[2];
    for (int k = 0; k < DH; ++k) {
        const float x = (float)yp[k];
        a0 += x * w[k];
        a1 += x * w[DH + k];
        a2 += x * w[2 * DH + k];
    }
    out[p * 3 + 0] = 1.f / (1.f + expf(-a0));
    out[p * 3 + 1] = 1.f / (1.f + expf(-a1));
    out[p * 3 + 2] = 1.f / (1.f + expf(-a2));
}

// ---------------------------------------------------------------------------
extern "C" void kernel_launch(void* const* d_in, const int* in_sizes, int n_in,
                              void* d_out, int out_size, void* d_ws, size_t ws_size,
                              hipStream_t stream) {
    (void)in_sizes; (void)n_in; (void)out_size; (void)ws_size;
    const float* points  = (const float*)d_in[0];
    const float* normals = (const float*)d_in[1];
    const float* viewd   = (const float*)d_in[2];
    const float* fv      = (const float*)d_in[3];
    const float* phys    = (const float*)d_in[4];
    const float* rays    = (const float*)d_in[5];
    const float* cam     = (const float*)d_in[6];
    const float *v[5], *g[5], *b[5];
    for (int l = 0; l < 5; ++l) {
        v[l] = (const float*)d_in[7 + 3 * l];
        g[l] = (const float*)d_in[8 + 3 * l];
        b[l] = (const float*)d_in[9 + 3 * l];
    }

    // Workspace layout (f16 elements). Y2 aliases X0 (X dead after layer 0).
    _Float16* X0 = (_Float16*)d_ws;
    _Float16* Y1 = X0 + (size_t)P_TOTAL * D0PAD;
    _Float16* Y2 = X0;
    _Float16* W0 = Y1 + (size_t)P_TOTAL * DH;
    _Float16* W1 = W0 + (size_t)DH * D0PAD;
    _Float16* W2 = W1 + (size_t)DH * DH;
    _Float16* W3 = W2 + (size_t)DH * DH;
    float*    W4 = (float*)(W3 + (size_t)DH * DH);

    prep_w_f16<<<DH, 256, 0, stream>>>(v[0], g[0], W0, D0, D0PAD);
    prep_w_f16<<<DH, 256, 0, stream>>>(v[1], g[1], W1, DH, DH);
    prep_w_f16<<<DH, 256, 0, stream>>>(v[2], g[2], W2, DH, DH);
    prep_w_f16<<<DH, 256, 0, stream>>>(v[3], g[3], W3, DH, DH);
    prep_w_f32<<<3,  256, 0, stream>>>(v[4], g[4], W4, DH);

    features_kernel<<<P_TOTAL / 256, 256, 0, stream>>>(points, normals, viewd, fv,
                                                       phys, rays, cam, X0);

    dim3 gg(P_TOTAL / 256, DH / 64, 1);
    gemm_wmma_f16<<<gg, 256, 0, stream>>>(X0, D0PAD, W0, D0PAD, b[0], Y1, DH, D0PAD);
    gemm_wmma_f16<<<gg, 256, 0, stream>>>(Y1, DH,    W1, DH,    b[1], Y2, DH, DH);
    gemm_wmma_f16<<<gg, 256, 0, stream>>>(Y2, DH,    W2, DH,    b[2], Y1, DH, DH);
    gemm_wmma_f16<<<gg, 256, 0, stream>>>(Y1, DH,    W3, DH,    b[3], Y2, DH, DH);

    final_layer<<<P_TOTAL / 256, 256, 0, stream>>>(Y2, W4, b[4], (float*)d_out);
}
